// AttentionMemoryBank_89730456748488
// MI455X (gfx1250) — matrix operand order repaired and checked by reading, not compiled, not run on hardware
//
#include <hip/hip_runtime.h>

// ---------------------------------------------------------------------------
// AttentionMemoryBank forward for MI455X (gfx1250, wave32, WMMA bf16)
//   x:    (8, 256, 64, 64) f32      -> xf (B=8, K=256, N=4096)
//   mask: (8, 1, 64, 64) bool       -> forward value independent of mask
//   mem:  (1, 256, 1024) f32        -> m (K=256, M=1024)
//   logits = m^T @ x / 16 ; attn = softmax_M(logits) ; out = m @ attn
//   d_out = [ out (8*256*4096 f32) | logits (8*1024*4096 f32) ]
// Memory-bound (~200MB traffic @ 23.3 TB/s); bf16 WMMA w/ f32 accum.
// ---------------------------------------------------------------------------

typedef __attribute__((ext_vector_type(16))) __bf16 v16bf;
typedef __attribute__((ext_vector_type(8)))  __bf16 v8bf;
typedef __attribute__((ext_vector_type(8)))  float  v8f;
typedef __attribute__((ext_vector_type(4)))  float  v4f;

#define B_  8
#define K_  256
#define M_  1024
#define N_  4096
#define NT  64        // output columns per workgroup
#define XT_STRIDE 264 // bf16 elems: 528B row stride -> 4-bank lane stride, conflict-free b128
#define CB_STRIDE 68  // f32 elems: chunkbuf row padding
#define AT_STRIDE 40  // bf16 elems: 80B row stride, conflict-free b128

__device__ __forceinline__ __bf16 f2bf(float f) {
  unsigned u = __builtin_bit_cast(unsigned, f);
  u += 0x7FFFu + ((u >> 16) & 1u);              // round-to-nearest-even
  unsigned short h = (unsigned short)(u >> 16);
  return __builtin_bit_cast(__bf16, h);
}

union Frag { v8bf h[2]; v16bf v; };

// A-matrix 16x32 bf16: lane<16 holds K {0..15}; lane>=16 {16..31}
// caller pre-offsets p by (lane>=16 ? 16 : 0); halves at +0 and +8 elems.
__device__ __forceinline__ v16bf load_a_frag(const __bf16* p) {
  Frag f; f.h[0] = *(const v8bf*)p; f.h[1] = *(const v8bf*)(p + 8); return f.v;
}
// B-matrix 32x16 bf16: lane holds column lane%16, contiguous K run of 16
// (lane>=16 offset by +16 K, applied by caller); halves at +0 and +8 elems.
__device__ __forceinline__ v16bf load_b_frag(const __bf16* p) {
  Frag f; f.h[0] = *(const v8bf*)p; f.h[1] = *(const v8bf*)(p + 8); return f.v;
}

// ---------------------------------------------------------------------------
// Kernel 0: mem f32 (K x M) -> bf16 memA (K x M) and memT (M x K)
// ---------------------------------------------------------------------------
__global__ void __launch_bounds__(256)
cvt_mem(const float* __restrict__ mem, __bf16* __restrict__ memA,
        __bf16* __restrict__ memT) {
  int idx = blockIdx.x * 256 + threadIdx.x;     // grid = 1024 -> 262144 elems
  int m = idx & (M_ - 1);
  int k = idx >> 10;
  __bf16 v = f2bf(mem[idx]);
  memA[idx] = v;
  memT[(size_t)m * K_ + k] = v;
}

// ---------------------------------------------------------------------------
// Kernel 1: logits = memT @ x / 16, + online softmax stats per column
// One workgroup per (b, 64-col tile). 8 waves; M chunks of 128 (16 rows/wave).
// ---------------------------------------------------------------------------
__global__ void __launch_bounds__(256)
p1_logits(const float* __restrict__ x, const __bf16* __restrict__ memT,
          float* __restrict__ logits, float* __restrict__ colmax,
          float* __restrict__ colsum) {
  extern __shared__ char smem[];
  __bf16* xT  = (__bf16*)smem;                                // 64 x 264 bf16
  float* cb   = (float*)(smem + (size_t)64 * XT_STRIDE * 2);  // 128 x 68 f32
  float* pmax = cb + 128 * CB_STRIDE;                         // 4 x 64
  float* psum = pmax + 256;                                   // 4 x 64
  float* rmax = psum + 256;                                   // 64 running max
  float* rsum = rmax + 64;                                    // 64 running sumexp
  float* nmax = rsum + 64;                                    // 64 chunk new-max

  const int tid   = threadIdx.x;
  const int b     = blockIdx.y;
  const int n0    = blockIdx.x * NT;
  const int wv    = tid >> 5;
  const int lane  = tid & 31;
  const int lrow  = lane & 15;
  const int lhalf = lane >> 4;

  // Stage x tile (K_ x 64 f32) -> transposed bf16 xT[n][k]
  {
    const int n  = tid & 63;
    const int kb = tid >> 6;
    const float* xb = x + (size_t)b * K_ * N_ + n0 + n;
    for (int i = 0; i < 64; ++i) {
      int k = i * 4 + kb;
      xT[n * XT_STRIDE + k] = f2bf(xb[(size_t)k * N_]);
    }
  }
  if (tid < 64) { rmax[tid] = -3.0e38f; rsum[tid] = 0.0f; }
  __syncthreads();

  const __bf16* arowbase = memT + (size_t)(wv * 16 + lrow) * K_;
  const int tc = tid & 63;
  const int tr = tid >> 6;

  for (int mc = 0; mc < 8; ++mc) {
    const int mbase = mc * 128;
    v8f acc[4] = {};
    const __bf16* arow = arowbase + (size_t)mbase * K_;
    #pragma unroll
    for (int kt = 0; kt < 8; ++kt) {
      v16bf a = load_a_frag(arow + kt * 32 + lhalf * 16);
      const __bf16* bbase = xT + lrow * XT_STRIDE + kt * 32 + lhalf * 16;
      #pragma unroll
      for (int nt2 = 0; nt2 < 4; ++nt2) {
        v16bf bf = load_b_frag(bbase + nt2 * 16 * XT_STRIDE);
        acc[nt2] = __builtin_amdgcn_wmma_f32_16x16x32_bf16(
            false, a, false, bf, (short)0, acc[nt2], false, false);
      }
    }
    // C layout: VGPR j, lane L -> row = j + 8*(L>=16), col = L%16
    #pragma unroll
    for (int nt2 = 0; nt2 < 4; ++nt2)
      #pragma unroll
      for (int j = 0; j < 8; ++j)
        cb[(wv * 16 + j + 8 * lhalf) * CB_STRIDE + nt2 * 16 + lrow] =
            acc[nt2][j] * 0.0625f;                     // 1/sqrt(K)=1/16
    __syncthreads();

    // per-column partial max over this 128-row chunk
    float pm = -3.0e38f;
    for (int r = tr; r < 128; r += 4) pm = fmaxf(pm, cb[r * CB_STRIDE + tc]);
    pmax[tr * 64 + tc] = pm;
    __syncthreads();
    if (tid < 64) {
      float nm = fmaxf(fmaxf(pmax[tid], pmax[64 + tid]),
                       fmaxf(pmax[128 + tid], pmax[192 + tid]));
      nmax[tid] = fmaxf(nm, rmax[tid]);
    }
    __syncthreads();
    {
      float nm = nmax[tc];
      float ps = 0.0f;
      for (int r = tr; r < 128; r += 4) ps += __expf(cb[r * CB_STRIDE + tc] - nm);
      psum[tr * 64 + tc] = ps;
    }
    // write logits chunk to global (float4, coalesced)
    {
      const int c4 = (tid & 15) * 4;
      const int r0 = tid >> 4;
      float* lg = logits + ((size_t)b * M_ + mbase) * N_ + n0 + c4;
      for (int r = r0; r < 128; r += 16)
        *(v4f*)(lg + (size_t)r * N_) = *(const v4f*)&cb[r * CB_STRIDE + c4];
    }
    __syncthreads();
    if (tid < 64) {           // online rescale of running sum
      float nm = nmax[tid];
      float s = psum[tid] + psum[64 + tid] + psum[128 + tid] + psum[192 + tid];
      rsum[tid] = rsum[tid] * __expf(rmax[tid] - nm) + s;
      rmax[tid] = nm;
    }
    __syncthreads();
  }
  if (tid < 64) {
    const size_t sidx = ((size_t)b * 64 + blockIdx.x) * 64 + tid;
    colmax[sidx] = rmax[tid];
    colsum[sidx] = rsum[tid];
  }
}

// ---------------------------------------------------------------------------
// Kernel 2: out = memA @ softmax(logits).  M chunks of 32; attn staged bf16.
// ---------------------------------------------------------------------------
__global__ void __launch_bounds__(256)
p2_out(const float* __restrict__ logits, const __bf16* __restrict__ memA,
       const float* __restrict__ colmax, const float* __restrict__ colsum,
       float* __restrict__ out) {
  __shared__ __attribute__((aligned(16))) __bf16 attnT[64 * AT_STRIDE];
  __shared__ float cmax[64];
  __shared__ float cinv[64];

  const int tid   = threadIdx.x;
  const int b     = blockIdx.y;
  const int n0    = blockIdx.x * NT;
  const int wv    = tid >> 5;
  const int lane  = tid & 31;
  const int lrow  = lane & 15;
  const int lhalf = lane >> 4;

  if (tid < 64) {
    size_t sidx = ((size_t)b * 64 + blockIdx.x) * 64 + tid;
    cmax[tid] = colmax[sidx];
    cinv[tid] = 1.0f / colsum[sidx];
  }
  __syncthreads();

  v8f acc[2][4] = {};
  const int tc = tid & 63;
  const int tr = tid >> 6;

  for (int mc = 0; mc < 32; ++mc) {
    // stage attn chunk (32 m-rows x 64 n-cols) transposed: attnT[n][m]
    {
      const float* lg = logits + ((size_t)b * M_ + mc * 32) * N_ + n0 + tc;
      float cm = cmax[tc], ci = cinv[tc];
      for (int r = tr; r < 32; r += 4) {
        float w = __expf(lg[(size_t)r * N_] - cm) * ci;
        attnT[tc * AT_STRIDE + r] = f2bf(w);
      }
    }
    __syncthreads();
    #pragma unroll
    for (int rt = 0; rt < 2; ++rt) {
      const __bf16* arow = memA + (size_t)(wv * 32 + rt * 16 + lrow) * M_ +
                           mc * 32 + lhalf * 16;
      v16bf a = load_a_frag(arow);
      #pragma unroll
      for (int nt2 = 0; nt2 < 4; ++nt2) {
        v16bf bf = load_b_frag(attnT + (nt2 * 16 + lrow) * AT_STRIDE + lhalf * 16);
        acc[rt][nt2] = __builtin_amdgcn_wmma_f32_16x16x32_bf16(
            false, a, false, bf, (short)0, acc[rt][nt2], false, false);
      }
    }
    __syncthreads();   // protect attnT reuse (WAR)
  }

  #pragma unroll
  for (int rt = 0; rt < 2; ++rt)
    #pragma unroll
    for (int nt2 = 0; nt2 < 4; ++nt2)
      #pragma unroll
      for (int j = 0; j < 8; ++j)
        out[((size_t)b * K_ + wv * 32 + rt * 16 + j + 8 * lhalf) * N_ +
            n0 + nt2 * 16 + lrow] = acc[rt][nt2][j];
}

// ---------------------------------------------------------------------------
extern "C" void kernel_launch(void* const* d_in, const int* in_sizes, int n_in,
                              void* d_out, int out_size, void* d_ws, size_t ws_size,
                              hipStream_t stream) {
  const float* x   = (const float*)d_in[0];
  // d_in[1] = mask: forward output is independent of it (blend is identity in value)
  const float* mem = (const float*)d_in[2];

  char* ws = (char*)d_ws;
  __bf16* memA  = (__bf16*)ws;                                  // 512 KB
  __bf16* memT  = (__bf16*)(ws + (size_t)512 * 1024);           // 512 KB
  float*  cmaxg = (float*)(ws + (size_t)1024 * 1024);           // 128 KB
  float*  csumg = (float*)(ws + (size_t)1024 * 1024 + 131072);  // 128 KB

  float* out    = (float*)d_out;                 // 8*256*4096 f32
  float* logits = out + (size_t)B_ * K_ * N_;    // 8*1024*4096 f32

  cvt_mem<<<1024, 256, 0, stream>>>(mem, memA, memT);

  const size_t p1_lds = (size_t)64 * XT_STRIDE * 2      // xT
                      + (size_t)128 * CB_STRIDE * 4     // chunkbuf
                      + (256 + 256 + 64 + 64 + 64) * 4; // reductions
  p1_logits<<<dim3(64, 8), 256, p1_lds, stream>>>(x, memT, logits, cmaxg, csumg);
  p2_out<<<dim3(64, 8), 256, 0, stream>>>(logits, memA, cmaxg, csumg, out);
}
